// MerchantGraphEncoder_55525337202746
// MI455X (gfx1250) — matrix operand order repaired and checked by reading, not compiled, not run on hardware
//
#include <hip/hip_runtime.h>

// ---------------------------------------------------------------------------
// 3-layer GCN for MI455X (gfx1250, wave32, WMMA).
// GEMMs run as bf16 WMMA (v_wmma_f32_16x16x32_bf16, fp32 accumulate) with the
// weight matrix staged in LDS (320KB/WGP on CDNA5) so B reads ride DScnt while
// A streams on LOADcnt; message passing / pooling stay fp32 global atomics.
// ---------------------------------------------------------------------------

typedef __attribute__((ext_vector_type(16))) __bf16 bf16x16;
typedef __attribute__((ext_vector_type(8)))  __bf16 bf16x8;
typedef __attribute__((ext_vector_type(8)))  float  f32x8;

#define IN_DIM 128
#define HID    256
#define NGRAPH 64

union Frag16 { bf16x16 v; bf16x8 h[2]; };

// ----------------------------- norm setup ---------------------------------

__global__ void k_deg_init(float* __restrict__ deg, int n) {
    int i = blockIdx.x * blockDim.x + threadIdx.x;
    if (i < n) deg[i] = 1.0f;   // self loop contributes 1 to every degree
}

__global__ void k_deg_edges(const int* __restrict__ dst, float* __restrict__ deg, int e) {
    int i = blockIdx.x * blockDim.x + threadIdx.x;
    if (i < e) atomicAdd(&deg[dst[i]], 1.0f);
}

__global__ void k_dis(float* __restrict__ d, int n) {
    int i = blockIdx.x * blockDim.x + threadIdx.x;
    if (i < n) d[i] = rsqrtf(d[i]);   // deg >= 1 always (self loops)
}

// --------------------------- bf16 conversion ------------------------------

__global__ void k_cvt_bf16(const float* __restrict__ in, __bf16* __restrict__ out, long n) {
    long i = (long)blockIdx.x * blockDim.x + threadIdx.x;
    if (i < n) out[i] = (__bf16)in[i];
}

// W[k][n] (KxN row-major fp32) -> Wt[n][k] (NxK row-major bf16)
__global__ void k_transpose_bf16(const float* __restrict__ W, __bf16* __restrict__ Wt,
                                 int K, int Nc) {
    int i = blockIdx.x * blockDim.x + threadIdx.x;
    if (i >= K * Nc) return;
    int n = i / K;
    int k = i - n * K;
    Wt[i] = (__bf16)W[(long)k * Nc + n];
}

// ------------------------------ WMMA GEMM ---------------------------------
// C[M][HID] fp32 = A[M][K] bf16 * Bt[HID][K]^T bf16, K in {128, 256}.
// Block = 256 threads = 8 waves. Wave w: M-tile = 2*blockIdx.x + (w>>2),
// N-strip = (w&3)*64 (4 accumulators of 16 columns each). K stepped by 32.
//
// Bt (<=128KB bf16) is staged once per block into LDS (CDNA5: 320KB/WGP, so 2
// blocks/WGP still fit): B fragment reads become ds_load_b128 on DScnt while A
// streams from global on LOADcnt behind a WGP-scope prefetch — the two counters
// let WMMAs retire without a full VMEM flush.
//
// Fragment layouts used here (contiguous k, same order for A and B):
//  A (16x32 bf16):  lane<16 -> M=lane,    K=k0..k0+15
//                   lane>=16-> M=lane-16, K=k0+16..k0+31
//  B (32x16 bf16):  lane%16 = N column;   lane<16 -> K=k0..k0+15,
//                                         lane>=16-> K=k0+16..k0+31
//  C/D (16x16 f32): VGPR g, lane<16 -> (M=g,   N=lane)
//                           lane>=16-> (M=g+8, N=lane-16)
__global__ void __launch_bounds__(256)
k_gemm_bf16(const __bf16* __restrict__ A, const __bf16* __restrict__ Bt,
            float* __restrict__ C, int M, int K) {
    __shared__ __bf16 sB[HID * HID];          // 128 KB, holds Bt[HID][K]

    // ---- cooperative fill: Bt -> LDS in 16B chunks (before any exit!) ----
    const int tid   = threadIdx.x;
    const int total = (HID * K) / 8;          // number of 8-elem chunks
    for (int c = tid; c < total; c += 256) {
        *(bf16x8*)(sB + (size_t)c * 8) = *(const bf16x8*)(Bt + (size_t)c * 8);
    }
    __syncthreads();                          // s_barrier_signal/wait

    const int lane = tid & 31;
    const int wave = tid >> 5;
    const int m0   = (blockIdx.x * 2 + (wave >> 2)) * 16;
    const int n0   = (wave & 3) * 64;
    if (m0 >= M) return;

    const int half = lane >> 4;
    const int r    = lane & 15;

    const __bf16* pa  = A  + (size_t)(m0 + r) * K + half * 16;
    const __bf16* sb0 = sB + (size_t)(n0 +  0 + r) * K + half * 16;
    const __bf16* sb1 = sB + (size_t)(n0 + 16 + r) * K + half * 16;
    const __bf16* sb2 = sB + (size_t)(n0 + 32 + r) * K + half * 16;
    const __bf16* sb3 = sB + (size_t)(n0 + 48 + r) * K + half * 16;

    f32x8 acc0 = {}, acc1 = {}, acc2 = {}, acc3 = {};

    for (int k0 = 0; k0 < K; k0 += 32) {
        // keep the next A tile marching into the near caches
        if (k0 + 32 < K) __builtin_prefetch(pa + k0 + 32, 0, 3);

        Frag16 a, bf0, bf1, bf2, bf3;
        a.h[0]   = *(const bf16x8*)(pa  + k0);        // global, LOADcnt
        a.h[1]   = *(const bf16x8*)(pa  + k0 + 8);
        bf0.h[0] = *(const bf16x8*)(sb0 + k0);        // LDS, DScnt
        bf0.h[1] = *(const bf16x8*)(sb0 + k0 + 8);
        bf1.h[0] = *(const bf16x8*)(sb1 + k0);
        bf1.h[1] = *(const bf16x8*)(sb1 + k0 + 8);
        bf2.h[0] = *(const bf16x8*)(sb2 + k0);
        bf2.h[1] = *(const bf16x8*)(sb2 + k0 + 8);
        bf3.h[0] = *(const bf16x8*)(sb3 + k0);
        bf3.h[1] = *(const bf16x8*)(sb3 + k0 + 8);

        acc0 = __builtin_amdgcn_wmma_f32_16x16x32_bf16(false, a.v, false, bf0.v,
                                                       (short)0, acc0, false, false);
        acc1 = __builtin_amdgcn_wmma_f32_16x16x32_bf16(false, a.v, false, bf1.v,
                                                       (short)0, acc1, false, false);
        acc2 = __builtin_amdgcn_wmma_f32_16x16x32_bf16(false, a.v, false, bf2.v,
                                                       (short)0, acc2, false, false);
        acc3 = __builtin_amdgcn_wmma_f32_16x16x32_bf16(false, a.v, false, bf3.v,
                                                       (short)0, acc3, false, false);
    }

#pragma unroll
    for (int g = 0; g < 8; ++g) {
        float* row = C + (size_t)(m0 + g + half * 8) * HID + (n0 + r);
        row[0]  = acc0[g];
        row[16] = acc1[g];
        row[32] = acc2[g];
        row[48] = acc3[g];
    }
}

// --------------------------- message passing ------------------------------
// One thread per (edge, 8-feature chunk): agg[dst] += h[src] * dis[src]*dis[dst]
__global__ void k_edge_scatter(const int* __restrict__ src, const int* __restrict__ dst,
                               const float* __restrict__ dis, const float* __restrict__ h,
                               float* __restrict__ agg, int E) {
    int t = blockIdx.x * blockDim.x + threadIdx.x;
    int e  = t >> 5;          // HID/8 = 32 chunks per edge
    int f0 = (t & 31) * 8;
    if (e >= E) return;
    int s = src[e], d = dst[e];
    float nrm = dis[s] * dis[d];
    const float* hp = h   + (size_t)s * HID + f0;
    float*       ap = agg + (size_t)d * HID + f0;
#pragma unroll
    for (int i = 0; i < 8; ++i) atomicAdd(ap + i, hp[i] * nrm);
}

// self-loop term + bias + ReLU, cast to bf16 for the next layer's GEMM
__global__ void k_post_relu(const float* __restrict__ agg, const float* __restrict__ h,
                            const float* __restrict__ dis, const float* __restrict__ b,
                            __bf16* __restrict__ out, int Nn) {
    long t = (long)blockIdx.x * blockDim.x + threadIdx.x;
    if (t >= (long)Nn * HID) return;
    int i = (int)(t >> 8);
    int f = (int)(t & (HID - 1));
    float di = dis[i];
    float v  = agg[t] + h[t] * di * di + b[f];
    out[t] = (__bf16)fmaxf(v, 0.0f);
}

// layer-2 epilogue fused with mean-pool accumulation
__global__ void k_pool(const float* __restrict__ agg, const float* __restrict__ h,
                       const float* __restrict__ dis, const float* __restrict__ b2,
                       const int* __restrict__ batch, float* __restrict__ out,
                       float* __restrict__ cnt, int Nn) {
    long t = (long)blockIdx.x * blockDim.x + threadIdx.x;
    if (t >= (long)Nn * HID) return;
    int i = (int)(t >> 8);
    int f = (int)(t & (HID - 1));
    float di = dis[i];
    float v  = agg[t] + h[t] * di * di + b2[f];
    int g = batch[i];
    atomicAdd(&out[(size_t)g * HID + f], v);
    if (f == 0) atomicAdd(&cnt[g], 1.0f);
}

__global__ void k_div(float* __restrict__ out, const float* __restrict__ cnt, int n) {
    int t = blockIdx.x * blockDim.x + threadIdx.x;
    if (t < n) out[t] /= fmaxf(cnt[t >> 8], 1.0f);
}

// ------------------------------- launcher ---------------------------------

extern "C" void kernel_launch(void* const* d_in, const int* in_sizes, int n_in,
                              void* d_out, int out_size, void* d_ws, size_t ws_size,
                              hipStream_t stream) {
    const float* x     = (const float*)d_in[0];
    const int*   ei    = (const int*)d_in[1];
    const int*   batch = (const int*)d_in[2];
    const float* W0 = (const float*)d_in[4];
    const float* b0 = (const float*)d_in[5];
    const float* W1 = (const float*)d_in[6];
    const float* b1 = (const float*)d_in[7];
    const float* W2 = (const float*)d_in[8];
    const float* b2 = (const float*)d_in[9];

    const int N = in_sizes[0] / IN_DIM;   // 100000
    const int E = in_sizes[1] / 2;        // 320000
    const int* src = ei;
    const int* dst = ei + E;

    // ---- carve workspace (256B aligned) ----
    char* p = (char*)d_ws;
    auto carve = [&](size_t bytes) -> void* {
        void* r = (void*)p;
        p += (bytes + 255) & ~(size_t)255;
        return r;
    };
    float*  dis  = (float*)carve((size_t)N * 4);
    __bf16* xb   = (__bf16*)carve((size_t)N * IN_DIM * 2);
    __bf16* inb  = (__bf16*)carve((size_t)N * HID * 2);
    float*  h    = (float*)carve((size_t)N * HID * 4);
    float*  agg  = (float*)carve((size_t)N * HID * 4);
    __bf16* W0t  = (__bf16*)carve((size_t)IN_DIM * HID * 2);
    __bf16* W1t  = (__bf16*)carve((size_t)HID * HID * 2);
    __bf16* W2t  = (__bf16*)carve((size_t)HID * HID * 2);
    float*  cnt  = (float*)carve((size_t)NGRAPH * 4);

    const int T = 256;
    const long NH = (long)N * HID;
    dim3 gN((N + T - 1) / T), gE((E + T - 1) / T);
    dim3 gNH((unsigned)((NH + T - 1) / T));
    dim3 gScat((unsigned)(((long)E * 32 + T - 1) / T));
    dim3 gGemm((unsigned)((N + 31) / 32));          // 2 M-tiles (32 rows) per block

    // ---- GCN normalization ----
    k_deg_init<<<gN, T, 0, stream>>>(dis, N);
    k_deg_edges<<<gE, T, 0, stream>>>(dst, dis, E);
    k_dis<<<gN, T, 0, stream>>>(dis, N);

    // ---- bf16 casts ----
    k_cvt_bf16<<<dim3((unsigned)(((long)N * IN_DIM + T - 1) / T)), T, 0, stream>>>(
        x, xb, (long)N * IN_DIM);
    k_transpose_bf16<<<(IN_DIM * HID + T - 1) / T, T, 0, stream>>>(W0, W0t, IN_DIM, HID);
    k_transpose_bf16<<<(HID * HID + T - 1) / T, T, 0, stream>>>(W1, W1t, HID, HID);
    k_transpose_bf16<<<(HID * HID + T - 1) / T, T, 0, stream>>>(W2, W2t, HID, HID);

    // ---- layer 0 ----
    hipMemsetAsync(agg, 0, (size_t)NH * 4, stream);
    k_gemm_bf16<<<gGemm, T, 0, stream>>>(xb, W0t, h, N, IN_DIM);
    k_edge_scatter<<<gScat, T, 0, stream>>>(src, dst, dis, h, agg, E);
    k_post_relu<<<gNH, T, 0, stream>>>(agg, h, dis, b0, inb, N);

    // ---- layer 1 ----
    hipMemsetAsync(agg, 0, (size_t)NH * 4, stream);
    k_gemm_bf16<<<gGemm, T, 0, stream>>>(inb, W1t, h, N, HID);
    k_edge_scatter<<<gScat, T, 0, stream>>>(src, dst, dis, h, agg, E);
    k_post_relu<<<gNH, T, 0, stream>>>(agg, h, dis, b1, inb, N);

    // ---- layer 2 + mean pool ----
    hipMemsetAsync(agg, 0, (size_t)NH * 4, stream);
    k_gemm_bf16<<<gGemm, T, 0, stream>>>(inb, W2t, h, N, HID);
    k_edge_scatter<<<gScat, T, 0, stream>>>(src, dst, dis, h, agg, E);

    hipMemsetAsync(d_out, 0, (size_t)out_size * 4, stream);
    hipMemsetAsync(cnt, 0, (size_t)NGRAPH * 4, stream);
    k_pool<<<gNH, T, 0, stream>>>(agg, h, dis, b2, batch, (float*)d_out, cnt, N);
    k_div<<<(out_size + T - 1) / T, T, 0, stream>>>((float*)d_out, cnt, out_size);
}
